// VisualContrastAttention_32736240730200
// MI455X (gfx1250) — compile-verified, matrix operand order checked by reading, not run on hardware
//
#include <hip/hip_runtime.h>
#include <math.h>

typedef __attribute__((ext_vector_type(16))) _Float16 v16h;
typedef __attribute__((ext_vector_type(8)))  float    v8f;
typedef __attribute__((ext_vector_type(4)))  int      v4i;

typedef __attribute__((address_space(1))) v4i gv4i;  // global int4
typedef __attribute__((address_space(3))) v4i lv4i;  // LDS int4

#define HEADS 12
#define HD    64
#define NC    64
#define BB    32
#define NTOK  1024
#define CDIM  768
#define EPSN  1e-6f
#define SCALE 0.125f   // HD^-0.5

// ---- gfx1250 async global->LDS path (guarded; sync fallback) ---------------
#if defined(__has_builtin)
#  if __has_builtin(__builtin_amdgcn_global_load_async_to_lds_b128)
#    define HAVE_ASYNC_LDS 1
#  endif
#endif
#ifndef HAVE_ASYNC_LDS
#  define HAVE_ASYNC_LDS 0
#endif

#if HAVE_ASYNC_LDS
#  if __has_builtin(__builtin_amdgcn_s_wait_asynccnt)
#    define ASYNC_WAIT() __builtin_amdgcn_s_wait_asynccnt(0)
#  else
#    define ASYNC_WAIT() asm volatile("s_wait_asynccnt 0" ::: "memory")
#  endif
#else
#  define ASYNC_WAIT() ((void)0)
#endif

static __device__ __forceinline__ void copy16_g2l(const float* gsrc, float* ldst) {
#if HAVE_ASYNC_LDS
  __builtin_amdgcn_global_load_async_to_lds_b128((gv4i*)(v4i*)gsrc,
                                                 (lv4i*)(v4i*)ldst, 0, 0);
#else
  *(float4*)ldst = *(const float4*)gsrc;
#endif
}

static __device__ __forceinline__ v8f wmma16(v16h a, v16h b, v8f c) {
  return __builtin_amdgcn_wmma_f32_16x16x32_f16(false, a, false, b, (short)0, c,
                                                false, false);
}

// A/B fragment loader from f16 array laid out [major][k] (ld = k-stride of major).
// A-frag: major = row (M).  B-frag: major = column (N), element = B[k][n].
static __device__ __forceinline__ v16h ldfrag_h(const _Float16* src, int maj0,
                                                int k0, int ld) {
  int lane = threadIdx.x & 31;
  const _Float16* p = src + (maj0 + (lane & 15)) * ld + k0 + ((lane >> 4) << 3);
  v16h f;
#pragma unroll
  for (int j = 0; j < 8; ++j) f[j] = p[j];           // K = g*8 + j
#pragma unroll
  for (int j = 0; j < 8; ++j) f[8 + j] = p[16 + j];  // K = 16 + g*8 + j
  return f;
}

static __device__ __forceinline__ v16h ldfrag_f32(const float* src, int maj0,
                                                  int k0, int ld) {
  int lane = threadIdx.x & 31;
  const float* p = src + (maj0 + (lane & 15)) * ld + k0 + ((lane >> 4) << 3);
  v16h f;
#pragma unroll
  for (int j = 0; j < 8; ++j) f[j] = (_Float16)p[j];
#pragma unroll
  for (int j = 0; j < 8; ++j) f[8 + j] = (_Float16)p[16 + j];
  return f;
}

// Reductions across the 16 lanes of one wave-half (same C-frag row).
static __device__ __forceinline__ float red16_max(float x) {
#pragma unroll
  for (int m = 1; m <= 8; m <<= 1) x = fmaxf(x, __shfl_xor(x, m, 32));
  return x;
}
static __device__ __forceinline__ float red16_sum(float x) {
#pragma unroll
  for (int m = 1; m <= 8; m <<= 1) x += __shfl_xor(x, m, 32);
  return x;
}

// ---------------------------------------------------------------------------
// Kernel 1: QKV = X * Wqkv^T, scatter into q/k (b,h,n,d) f16 and vT (b,h,d,n).
// Block = 128 rows x 64 cols (8 waves of 16 rows). Weight panel (64x32 f32)
// staged per-block into LDS via async copy, double buffered, 8x reuse.
// Grid: 256 M-blocks x 36 col-strips = 9216 blocks.
// ---------------------------------------------------------------------------
__global__ __launch_bounds__(256) void k_qkv(const float* __restrict__ x,
                                             const float* __restrict__ w,
                                             _Float16* __restrict__ qh,
                                             _Float16* __restrict__ kh,
                                             _Float16* __restrict__ vT) {
  __shared__ __align__(16) float wlds[2][64][32];  // 16 KB
  int mb = blockIdx.x / 36, nstrip = blockIdx.x % 36;
  int wv = threadIdx.x >> 5;
  int row0 = mb * 128 + wv * 16;
  int col0 = nstrip * 64;

  auto stage = [&](int k0, int bufi) {
#pragma unroll
    for (int i = 0; i < 2; ++i) {
      int flat = (i * 256 + (int)threadIdx.x) * 4;  // float index in 64x32 panel
      int n = flat >> 5, kc = flat & 31;
      copy16_g2l(w + (col0 + n) * CDIM + k0 + kc, &wlds[bufi][n][kc]);
    }
  };

  v8f acc[4] = {};
  stage(0, 0);
  for (int ks = 0; ks < CDIM / 32; ++ks) {
    int k0 = ks * 32;
    ASYNC_WAIT();
    __syncthreads();
    if (ks + 1 < CDIM / 32) stage(k0 + 32, (ks + 1) & 1);
    if (k0 + 32 < CDIM)
      __builtin_prefetch(x + (row0 + (threadIdx.x & 15)) * CDIM + k0 + 32, 0, 1);
    v16h a = ldfrag_f32(x, row0, k0, CDIM);
    const float* wt = &wlds[ks & 1][0][0];
#pragma unroll
    for (int t = 0; t < 4; ++t) {
      v16h b = ldfrag_f32(wt, t * 16, 0, 32);
      acc[t] = wmma16(a, b, acc[t]);
    }
  }

  int lane = threadIdx.x & 31, g = lane >> 4, nl = lane & 15;
  int which = col0 / CDIM;               // 0=q 1=k 2=v
  int h = ((col0 - which * CDIM) >> 6);  // head (strip is 64-aligned)
#pragma unroll
  for (int t = 0; t < 4; ++t) {
    int d = t * 16 + nl;
#pragma unroll
    for (int r = 0; r < 8; ++r) {
      int m = row0 + r + 8 * g;
      int bi = m >> 10, tok = m & 1023;
      int bh = bi * HEADS + h;
      _Float16 val = (_Float16)acc[t][r];
      if (which == 0)      qh[(bh * NTOK + tok) * HD + d] = val;
      else if (which == 1) kh[(bh * NTOK + tok) * HD + d] = val;
      else                 vT[(bh * HD + d) * NTOK + tok] = val;
    }
  }
}

// ---------------------------------------------------------------------------
// Kernel 2: landmark pooling (4x4 block mean of q) + pos/neg embeds -> f16.
// ---------------------------------------------------------------------------
__global__ __launch_bounds__(256) void k_landmark(const _Float16* __restrict__ qh,
                                                  const float* __restrict__ pep,
                                                  const float* __restrict__ pen,
                                                  _Float16* __restrict__ tpos,
                                                  _Float16* __restrict__ tneg) {
  int idx = blockIdx.x * blockDim.x + threadIdx.x;  // bh*4096 + p*64 + d
  int d = idx & 63;
  int p = (idx >> 6) & 63;
  int bh = idx >> 12;
  int py = p >> 3, px = p & 7;
  const _Float16* q = qh + bh * NTOK * HD;
  float s = 0.f;
#pragma unroll
  for (int iy = 0; iy < 4; ++iy)
#pragma unroll
    for (int ix = 0; ix < 4; ++ix) {
      int tok = (py * 4 + iy) * 32 + (px * 4 + ix);
      s += (float)q[tok * HD + d];
    }
  s *= (1.f / 16.f);
  tpos[idx] = (_Float16)(s + pep[p * HD + d]);
  tneg[idx] = (_Float16)(s + pen[p * HD + d]);
}

// ---------------------------------------------------------------------------
// Kernel 3: stage 1 flash attention (landmarks->tokens) for pos & neg,
// contrast + RMSNorm -> vcT (b,h,d,m) f16.  One block per (b,h).
// Waves 0-3: pos row-tiles 0..3;  waves 4-7: neg row-tiles 0..3.
// ---------------------------------------------------------------------------
__global__ __launch_bounds__(256) void k_stage1(const _Float16* __restrict__ tpos,
                                                const _Float16* __restrict__ tneg,
                                                const _Float16* __restrict__ kh,
                                                const _Float16* __restrict__ vT,
                                                const float* __restrict__ lam1,
                                                const float* __restrict__ w1,
                                                _Float16* __restrict__ vcT) {
  __shared__ _Float16 ptile[8][16][32];  // per-wave P staging (C-layout -> A-layout)
  __shared__ float obuf[2][NC][HD];      // v_hat pos / neg exchange
  int bh = blockIdx.x;
  int wv = threadIdx.x >> 5;
  int lane = threadIdx.x & 31, g = lane >> 4, nl = lane & 15;
  int rt = wv & 3, sign = wv >> 2;
  int row0 = rt * 16;
  const _Float16* t = (sign ? tneg : tpos) + bh * (NC * HD);
  const _Float16* kk = kh + bh * (NTOK * HD);
  const _Float16* vv = vT + bh * (NTOK * HD);
  v16h tA0 = ldfrag_h(t, row0, 0, HD);
  v16h tA1 = ldfrag_h(t, row0, 32, HD);
  float mrow[8], srow[8];
  v8f o[4] = {};
#pragma unroll
  for (int r = 0; r < 8; ++r) { mrow[r] = -1e30f; srow[r] = 0.f; }

  for (int n0 = 0; n0 < NTOK; n0 += 32) {
    if (n0 + 32 < NTOK)
      __builtin_prefetch(kk + (n0 + 32 + nl) * HD, 0, 1);
    v8f s0 = {}, s1 = {};
    {
      v16h b0 = ldfrag_h(kk, n0, 0, HD);
      v16h b1 = ldfrag_h(kk, n0, 32, HD);
      s0 = wmma16(tA0, b0, s0);
      s0 = wmma16(tA1, b1, s0);
      v16h b2 = ldfrag_h(kk, n0 + 16, 0, HD);
      v16h b3 = ldfrag_h(kk, n0 + 16, 32, HD);
      s1 = wmma16(tA0, b2, s1);
      s1 = wmma16(tA1, b3, s1);
    }
#pragma unroll
    for (int r = 0; r < 8; ++r) {
      float a = s0[r] * SCALE, b = s1[r] * SCALE;
      float mx = red16_max(fmaxf(a, b));
      float mnew = fmaxf(mrow[r], mx);
      float corr = __expf(mrow[r] - mnew);
      float p0 = __expf(a - mnew), p1 = __expf(b - mnew);
      srow[r] = srow[r] * corr + red16_sum(p0 + p1);
      mrow[r] = mnew;
#pragma unroll
      for (int t4 = 0; t4 < 4; ++t4) o[t4][r] *= corr;
      ptile[wv][r + 8 * g][nl] = (_Float16)p0;
      ptile[wv][r + 8 * g][16 + nl] = (_Float16)p1;
    }
    v16h pA = ldfrag_h(&ptile[wv][0][0], 0, 0, 32);
#pragma unroll
    for (int t4 = 0; t4 < 4; ++t4) {
      v16h vb = ldfrag_h(vv, t4 * 16, n0, NTOK);  // B[k=tok][n=d] = vT[d][tok]
      o[t4] = wmma16(pA, vb, o[t4]);
    }
  }
#pragma unroll
  for (int r = 0; r < 8; ++r) {
    float inv = 1.f / srow[r];
#pragma unroll
    for (int t4 = 0; t4 < 4; ++t4)
      obuf[sign][row0 + r + 8 * g][t4 * 16 + nl] = o[t4][r] * inv;
  }
  __syncthreads();

  // contrast + RMSNorm: thread -> (row = tid>>2, 16 dims)
  float l1 = lam1[0];
  int row = threadIdx.x >> 2;
  int d0 = (threadIdx.x & 3) * 16;
  float vals[16];
  float ss = 0.f;
#pragma unroll
  for (int j = 0; j < 16; ++j) {
    float v = obuf[0][row][d0 + j] - l1 * obuf[1][row][d0 + j];
    vals[j] = v;
    ss += v * v;
  }
  ss += __shfl_xor(ss, 1, 32);
  ss += __shfl_xor(ss, 2, 32);
  float inv = 0.5f / (sqrtf(ss * (1.f / 64.f)) + EPSN);  // * (1 - L1_INIT)
  _Float16* outp = vcT + bh * (NC * HD);
#pragma unroll
  for (int j = 0; j < 16; ++j) {
    int d = d0 + j;
    outp[d * NC + row] = (_Float16)(vals[j] * inv * w1[d]);  // transposed store
  }
}

// ---------------------------------------------------------------------------
// Kernel 4: stage 2 (tokens->landmarks), dual softmax over 64 landmarks,
// (p1 - l2*p2) @ v_contrast, RMSNorm, scatter into y (b,n,c) f16.
// ---------------------------------------------------------------------------
__global__ __launch_bounds__(256) void k_stage2(const _Float16* __restrict__ qh,
                                                const _Float16* __restrict__ tpos,
                                                const _Float16* __restrict__ tneg,
                                                const _Float16* __restrict__ vcT,
                                                const float* __restrict__ lam2,
                                                const float* __restrict__ w2,
                                                _Float16* __restrict__ y) {
  __shared__ _Float16 dtile[8][16][64];
  int bh = blockIdx.x;
  int b = bh / HEADS, h = bh - b * HEADS;
  int wv = threadIdx.x >> 5, lane = threadIdx.x & 31;
  int g = lane >> 4, nl = lane & 15;
  const _Float16* q = qh + bh * NTOK * HD;
  const _Float16* tp = tpos + bh * NC * HD;
  const _Float16* tn = tneg + bh * NC * HD;
  const _Float16* vc = vcT + bh * NC * HD;
  float l2 = lam2[0];
  float w2v[4];
#pragma unroll
  for (int t4 = 0; t4 < 4; ++t4) w2v[t4] = w2[t4 * 16 + nl];

  for (int i = 0; i < 8; ++i) {
    int row0 = (wv * 8 + i) * 16;
    v16h qA0 = ldfrag_h(q, row0, 0, HD);
    v16h qA1 = ldfrag_h(q, row0, 32, HD);
    v8f sp[4] = {}, sn[4] = {};
#pragma unroll
    for (int mt = 0; mt < 4; ++mt) {
      v16h bp0 = ldfrag_h(tp, mt * 16, 0, HD);   // B[k=d][n=m] = t[m][d]
      v16h bp1 = ldfrag_h(tp, mt * 16, 32, HD);
      sp[mt] = wmma16(qA0, bp0, sp[mt]);
      sp[mt] = wmma16(qA1, bp1, sp[mt]);
      v16h bn0 = ldfrag_h(tn, mt * 16, 0, HD);
      v16h bn1 = ldfrag_h(tn, mt * 16, 32, HD);
      sn[mt] = wmma16(qA0, bn0, sn[mt]);
      sn[mt] = wmma16(qA1, bn1, sn[mt]);
    }
#pragma unroll
    for (int r = 0; r < 8; ++r) {
      float mp = -1e30f, mn = -1e30f;
#pragma unroll
      for (int mt = 0; mt < 4; ++mt) {
        mp = fmaxf(mp, sp[mt][r] * SCALE);
        mn = fmaxf(mn, sn[mt][r] * SCALE);
      }
      mp = red16_max(mp);
      mn = red16_max(mn);
      float ep[4], en[4], sump = 0.f, sumn = 0.f;
#pragma unroll
      for (int mt = 0; mt < 4; ++mt) {
        ep[mt] = __expf(sp[mt][r] * SCALE - mp);
        en[mt] = __expf(sn[mt][r] * SCALE - mn);
        sump += ep[mt];
        sumn += en[mt];
      }
      sump = red16_sum(sump);
      sumn = red16_sum(sumn);
      float ip = 1.f / sump, inn = 1.f / sumn;
#pragma unroll
      for (int mt = 0; mt < 4; ++mt)
        dtile[wv][r + 8 * g][mt * 16 + nl] =
            (_Float16)(ep[mt] * ip - l2 * en[mt] * inn);
    }
    v16h dA0 = ldfrag_h(&dtile[wv][0][0], 0, 0, 64);
    v16h dA1 = ldfrag_h(&dtile[wv][0][0], 0, 32, 64);
    v8f o[4] = {};
#pragma unroll
    for (int t4 = 0; t4 < 4; ++t4) {
      v16h vb0 = ldfrag_h(vc, t4 * 16, 0, NC);   // B[k=m][n=d] = vcT[d][m]
      v16h vb1 = ldfrag_h(vc, t4 * 16, 32, NC);
      o[t4] = wmma16(dA0, vb0, o[t4]);
      o[t4] = wmma16(dA1, vb1, o[t4]);
    }
#pragma unroll
    for (int r = 0; r < 8; ++r) {
      float ss = 0.f;
#pragma unroll
      for (int t4 = 0; t4 < 4; ++t4) ss += o[t4][r] * o[t4][r];
      ss = red16_sum(ss);
      float inv = 0.5f / (sqrtf(ss * (1.f / 64.f)) + EPSN);  // * (1 - L2_INIT)
      int tok = row0 + r + 8 * g;
      _Float16* dst = y + (b * NTOK + tok) * CDIM + h * HD;
#pragma unroll
      for (int t4 = 0; t4 < 4; ++t4)
        dst[t4 * 16 + nl] = (_Float16)(o[t4][r] * inv * w2v[t4]);
    }
  }
}

// ---------------------------------------------------------------------------
// Kernel 5: OUT = Y * Wproj^T + b.  Same async-staged structure as k_qkv.
// Grid: 256 M-blocks x 12 col-strips = 3072 blocks.
// ---------------------------------------------------------------------------
__global__ __launch_bounds__(256) void k_proj(const _Float16* __restrict__ y,
                                              const float* __restrict__ w,
                                              const float* __restrict__ bias,
                                              float* __restrict__ out) {
  __shared__ __align__(16) float wlds[2][64][32];  // 16 KB
  int mb = blockIdx.x / 12, nstrip = blockIdx.x % 12;
  int wv = threadIdx.x >> 5;
  int row0 = mb * 128 + wv * 16;
  int col0 = nstrip * 64;

  auto stage = [&](int k0, int bufi) {
#pragma unroll
    for (int i = 0; i < 2; ++i) {
      int flat = (i * 256 + (int)threadIdx.x) * 4;
      int n = flat >> 5, kc = flat & 31;
      copy16_g2l(w + (col0 + n) * CDIM + k0 + kc, &wlds[bufi][n][kc]);
    }
  };

  v8f acc[4] = {};
  stage(0, 0);
  for (int ks = 0; ks < CDIM / 32; ++ks) {
    int k0 = ks * 32;
    ASYNC_WAIT();
    __syncthreads();
    if (ks + 1 < CDIM / 32) stage(k0 + 32, (ks + 1) & 1);
    if (k0 + 32 < CDIM)
      __builtin_prefetch(y + (row0 + (threadIdx.x & 15)) * CDIM + k0 + 32, 0, 1);
    v16h a = ldfrag_h(y, row0, k0, CDIM);
    const float* wt = &wlds[ks & 1][0][0];
#pragma unroll
    for (int t = 0; t < 4; ++t) {
      v16h b = ldfrag_f32(wt, t * 16, 0, 32);
      acc[t] = wmma16(a, b, acc[t]);
    }
  }

  int lane = threadIdx.x & 31, g = lane >> 4, nl = lane & 15;
#pragma unroll
  for (int t = 0; t < 4; ++t) {
    int n = col0 + t * 16 + nl;
    float bv = bias[n];
#pragma unroll
    for (int r = 0; r < 8; ++r) {
      int m = row0 + r + 8 * g;
      out[m * CDIM + n] = acc[t][r] + bv;
    }
  }
}

// ---------------------------------------------------------------------------
extern "C" void kernel_launch(void* const* d_in, const int* in_sizes, int n_in,
                              void* d_out, int out_size, void* d_ws, size_t ws_size,
                              hipStream_t stream) {
  (void)in_sizes; (void)n_in; (void)out_size; (void)ws_size;
  const float* x    = (const float*)d_in[0];
  const float* qkvw = (const float*)d_in[1];
  const float* pw   = (const float*)d_in[2];
  const float* pb   = (const float*)d_in[3];
  const float* pep  = (const float*)d_in[4];
  const float* pen  = (const float*)d_in[5];
  const float* l1   = (const float*)d_in[6];
  const float* l2   = (const float*)d_in[7];
  const float* w1   = (const float*)d_in[8];
  const float* w2   = (const float*)d_in[9];

  char* ws = (char*)d_ws;
  const size_t QKV = (size_t)BB * HEADS * NTOK * HD * sizeof(_Float16);  // 48 MB
  const size_t TSZ = (size_t)BB * HEADS * NC * HD * sizeof(_Float16);    // 3 MB
  _Float16* qh = (_Float16*)(ws);
  _Float16* kh = (_Float16*)(ws + QKV);
  _Float16* vT = (_Float16*)(ws + 2 * QKV);
  _Float16* yh = (_Float16*)(ws + 3 * QKV);
  _Float16* tp = (_Float16*)(ws + 4 * QKV);
  _Float16* tn = (_Float16*)(ws + 4 * QKV + TSZ);
  _Float16* vc = (_Float16*)(ws + 4 * QKV + 2 * TSZ);
  float* out = (float*)d_out;

  k_qkv<<<9216, 256, 0, stream>>>(x, qkvw, qh, kh, vT);
  k_landmark<<<6144, 256, 0, stream>>>(qh, pep, pen, tp, tn);
  k_stage1<<<384, 256, 0, stream>>>(tp, tn, kh, vT, l1, w1, vc);
  k_stage2<<<384, 256, 0, stream>>>(qh, tp, tn, vc, l2, w2, yh);
  k_proj<<<3072, 256, 0, stream>>>(yh, pw, pb, out);
}
